// GCNEncoder_67654324846925
// MI455X (gfx1250) — matrix operand order repaired and checked by reading, not compiled, not run on hardware
//
#include <hip/hip_runtime.h>

typedef __attribute__((ext_vector_type(16))) _Float16 v16h;
typedef __attribute__((ext_vector_type(8)))  float    v8f;

#define IN_C 256
#define HID  32
#define OUTC 16

// ---------------- degree / dinv ----------------

__global__ void k_init_deg(float* __restrict__ deg, int N) {
    int i = blockIdx.x * blockDim.x + threadIdx.x;
    if (i < N) deg[i] = 1.0f;  // self-loop
}

__global__ void k_count_deg(const long long* __restrict__ dst,
                            float* __restrict__ deg, long long E) {
    long long e = blockIdx.x * (long long)blockDim.x + threadIdx.x;
    if (e < E) atomicAdd(&deg[dst[e]], 1.0f);
}

__global__ void k_rsqrt(float* __restrict__ deg, int N) {
    int i = blockIdx.x * blockDim.x + threadIdx.x;
    if (i < N) deg[i] = rsqrtf(deg[i]);
}

// ---------------- weight pre-conversion into B-fragment order ----------------
// Fragment layout (16-bit B, 32xN tile, wave32):
//   lane L holds column n = (L&15) [+16*tile], elements j=0..15 hold K = (L>>4)*8 + (j&7) + 16*(j>>3).
// w1f linear index: ((kcIdx*2 + tile)*32 + lane)*16 + j   (8 k-chunks, 2 n-tiles)
__global__ void k_cvtW1(const float* __restrict__ W1, _Float16* __restrict__ w1f) {
    int t = blockIdx.x * blockDim.x + threadIdx.x;
    if (t >= 8 * 2 * 32 * 16) return;
    int j    = t & 15;
    int lane = (t >> 4) & 31;
    int tile = (t >> 9) & 1;
    int kc   = t >> 10;
    int n = (lane & 15) + 16 * tile;
    int k = kc * 32 + ((lane >> 4) << 3) + (j & 7) + ((j >> 3) << 4);
    w1f[t] = (_Float16)W1[k * HID + n];
}

// w2f linear index: lane*16 + j  (single k-chunk, single n-tile)
__global__ void k_cvtW2(const float* __restrict__ W2, _Float16* __restrict__ w2f) {
    int t = blockIdx.x * blockDim.x + threadIdx.x;
    if (t >= 32 * 16) return;
    int j    = t & 15;
    int lane = (t >> 4) & 31;
    int n = lane & 15;
    int k = ((lane >> 4) << 3) + (j & 7) + ((j >> 3) << 4);
    w2f[t] = (_Float16)W2[k * OUTC + n];
}

// ---------------- GEMM1: hw[N,32] = x[N,256] @ W1[256,32] (WMMA, full tiles) --

__global__ void k_gemm1(const float* __restrict__ x, const _Float16* __restrict__ w1f,
                        float* __restrict__ hw, int MB) {
    int gwave = (int)((blockIdx.x * (unsigned)blockDim.x + threadIdx.x) >> 5);
    int lane  = threadIdx.x & 31;
    if (gwave >= MB) return;                 // wave-uniform exit; EXEC all-1s inside
    int row0  = gwave << 4;
    int m     = lane & 15;                   // A row / D column
    int hi    = lane >> 4;
    int khalf = hi << 3;                     // A layout: K 0..7 / 8..15 (+16 block)

    const float* xr = x + (size_t)(row0 + m) * IN_C;
    __builtin_prefetch(xr, 0, 3);            // global_prefetch_b8

    v8f acc0 = {0.f,0.f,0.f,0.f,0.f,0.f,0.f,0.f};
    v8f acc1 = acc0;

    for (int kc = 0; kc < IN_C; kc += 32) {
        int kcIdx = kc >> 5;
        v16h a;
#pragma unroll
        for (int j = 0; j < 8; ++j) {
            a[j]     = (_Float16)xr[kc + khalf + j];        // K = khalf+j
            a[j + 8] = (_Float16)xr[kc + 16 + khalf + j];   // K = 16+khalf+j
        }
        v16h b0 = *(const v16h*)(w1f + (((size_t)kcIdx * 2 + 0) * 32 + lane) * 16);
        v16h b1 = *(const v16h*)(w1f + (((size_t)kcIdx * 2 + 1) * 32 + lane) * 16);
        acc0 = __builtin_amdgcn_wmma_f32_16x16x32_f16(false, a, false, b0,
                                                      (short)0, acc0, false, false);
        acc1 = __builtin_amdgcn_wmma_f32_16x16x32_f16(false, a, false, b1,
                                                      (short)0, acc1, false, false);
    }
#pragma unroll
    for (int r = 0; r < 8; ++r) {
        int row = row0 + khalf + r;          // D: VGPR r -> M = r (+8 for upper half)
        hw[(size_t)row * HID + m]      = acc0[r];
        hw[(size_t)row * HID + 16 + m] = acc1[r];
    }
}

// ---------------- GEMM2: h2w[N,16] = h[N,32] @ W2[32,16] (single WMMA) --------

__global__ void k_gemm2(const float* __restrict__ h, const _Float16* __restrict__ w2f,
                        float* __restrict__ h2w, int MB) {
    int gwave = (int)((blockIdx.x * (unsigned)blockDim.x + threadIdx.x) >> 5);
    int lane  = threadIdx.x & 31;
    if (gwave >= MB) return;
    int row0  = gwave << 4;
    int m     = lane & 15;
    int hi    = lane >> 4;
    int khalf = hi << 3;

    const float* hr = h + (size_t)(row0 + m) * HID;

    v16h a;
#pragma unroll
    for (int j = 0; j < 8; ++j) {
        a[j]     = (_Float16)hr[khalf + j];
        a[j + 8] = (_Float16)hr[16 + khalf + j];
    }
    v16h b = *(const v16h*)(w2f + (size_t)lane * 16);
    v8f acc = {0.f,0.f,0.f,0.f,0.f,0.f,0.f,0.f};
    acc = __builtin_amdgcn_wmma_f32_16x16x32_f16(false, a, false, b,
                                                 (short)0, acc, false, false);
#pragma unroll
    for (int r = 0; r < 8; ++r) {
        int row = row0 + khalf + r;
        h2w[(size_t)row * OUTC + m] = acc[r];
    }
}

// ---------------- scalar tail GEMM (only if N % 16 != 0; never for N=100000) --

__global__ void k_gemm_tail(const float* __restrict__ A, const float* __restrict__ W,
                            float* __restrict__ out, int row0, int N, int K, int NC) {
    int t = blockIdx.x * blockDim.x + threadIdx.x;
    int r = row0 + t / NC;
    int c = t % NC;
    if (r < N) {
        float s = 0.f;
        for (int k = 0; k < K; ++k) s += A[(size_t)r * K + k] * W[k * NC + c];
        out[(size_t)r * NC + c] = s;
    }
}

// ---------------- layer-1 aggregation ----------------

__global__ void k_self1(const float* __restrict__ hw, const float* __restrict__ dinv,
                        float* __restrict__ h, int N) {
    int t = blockIdx.x * blockDim.x + threadIdx.x;
    int i = t >> 5, c = t & 31;
    if (i < N) {
        float di = dinv[i];
        h[(size_t)i * HID + c] = hw[(size_t)i * HID + c] * di * di;
    }
}

__global__ void k_scatter1(const long long* __restrict__ src, const long long* __restrict__ dst,
                           const float* __restrict__ hw, const float* __restrict__ dinv,
                           float* __restrict__ h, long long E) {
    long long t = blockIdx.x * (long long)blockDim.x + threadIdx.x;
    long long e = t >> 5;
    int c = (int)(t & 31);
    if (e < E) {
        long long s = src[e], d = dst[e];
        float norm = dinv[s] * dinv[d];
        atomicAdd(&h[(size_t)d * HID + c], hw[(size_t)s * HID + c] * norm);
    }
}

__global__ void k_bias_relu1(float* __restrict__ h, const float* __restrict__ b1, int N) {
    int t = blockIdx.x * blockDim.x + threadIdx.x;
    int i = t >> 5, c = t & 31;
    if (i < N) {
        float v = h[(size_t)i * HID + c] + b1[c];
        h[(size_t)i * HID + c] = v > 0.f ? v : 0.f;
    }
}

// ---------------- layer-2 aggregation ----------------

__global__ void k_self2(const float* __restrict__ h2w, const float* __restrict__ dinv,
                        const float* __restrict__ b2, float* __restrict__ out, int N) {
    int t = blockIdx.x * blockDim.x + threadIdx.x;
    int i = t >> 4, c = t & 15;
    if (i < N) {
        float di = dinv[i];
        out[(size_t)i * OUTC + c] = h2w[(size_t)i * OUTC + c] * di * di + b2[c];
    }
}

__global__ void k_scatter2(const long long* __restrict__ src, const long long* __restrict__ dst,
                           const float* __restrict__ h2w, const float* __restrict__ dinv,
                           float* __restrict__ out, long long E) {
    long long t = blockIdx.x * (long long)blockDim.x + threadIdx.x;
    long long e = t >> 4;
    int c = (int)(t & 15);
    if (e < E) {
        long long s = src[e], d = dst[e];
        float norm = dinv[s] * dinv[d];
        atomicAdd(&out[(size_t)d * OUTC + c], h2w[(size_t)s * OUTC + c] * norm);
    }
}

// ---------------- launch ----------------

extern "C" void kernel_launch(void* const* d_in, const int* in_sizes, int n_in,
                              void* d_out, int out_size, void* d_ws, size_t ws_size,
                              hipStream_t stream) {
    const float*     x   = (const float*)d_in[0];
    const long long* ei  = (const long long*)d_in[1];   // int64 edge_index [2, E]
    const float*     W1  = (const float*)d_in[2];
    const float*     b1  = (const float*)d_in[3];
    const float*     W2  = (const float*)d_in[4];
    const float*     b2  = (const float*)d_in[5];
    float*           out = (float*)d_out;

    int       N = in_sizes[0] / IN_C;                   // 100000
    long long E = (long long)in_sizes[1] / 2;           // 3200000
    const long long* src = ei;
    const long long* dst = ei + E;

    // workspace: dinv[N] | hw[32N] | h[32N] | h2w[16N] | w1f[8192 f16] | w2f[512 f16]
    float* dinv = (float*)d_ws;
    float* hw   = dinv + N;
    float* h    = hw + (size_t)32 * N;
    float* h2w  = h  + (size_t)32 * N;
    _Float16* w1f = (_Float16*)(h2w + (size_t)16 * N);
    _Float16* w2f = w1f + 8 * 2 * 32 * 16;

    const int T = 256;
    int MB   = N >> 4;                                  // full 16-row WMMA tiles
    int tail = N & 15;

    // degree (with self-loops) -> dinv
    k_init_deg <<<(N + T - 1) / T, T, 0, stream>>>(dinv, N);
    k_count_deg<<<(unsigned)((E + T - 1) / T), T, 0, stream>>>(dst, dinv, E);
    k_rsqrt    <<<(N + T - 1) / T, T, 0, stream>>>(dinv, N);

    // weight fragments (f16, B-fragment lane order)
    k_cvtW1<<<(8 * 2 * 32 * 16 + T - 1) / T, T, 0, stream>>>(W1, w1f);
    k_cvtW2<<<(32 * 16 + T - 1) / T, T, 0, stream>>>(W2, w2f);

    // layer 1
    if (MB > 0)
        k_gemm1<<<(MB * 32 + T - 1) / T, T, 0, stream>>>(x, w1f, hw, MB);
    if (tail)
        k_gemm_tail<<<(tail * HID + T - 1) / T, T, 0, stream>>>(x, W1, hw, MB * 16, N, IN_C, HID);
    k_self1    <<<((size_t)N * 32 + T - 1) / T, T, 0, stream>>>(hw, dinv, h, N);
    k_scatter1 <<<(unsigned)((E * 32 + T - 1) / T), T, 0, stream>>>(src, dst, hw, dinv, h, E);
    k_bias_relu1<<<((size_t)N * 32 + T - 1) / T, T, 0, stream>>>(h, b1, N);

    // layer 2
    if (MB > 0)
        k_gemm2<<<(MB * 32 + T - 1) / T, T, 0, stream>>>(h, w2f, h2w, MB);
    if (tail)
        k_gemm_tail<<<(tail * OUTC + T - 1) / T, T, 0, stream>>>(h, W2, h2w, MB * 16, N, HID, OUTC);
    k_self2    <<<((size_t)N * 16 + T - 1) / T, T, 0, stream>>>(h2w, dinv, b2, out, N);
    k_scatter2 <<<(unsigned)((E * 16 + T - 1) / T), T, 0, stream>>>(src, dst, h2w, dinv, out, E);
}
